// TripletContrastiveLoss_69818988364136
// MI455X (gfx1250) — matrix operand order repaired and checked by reading, not compiled, not run on hardware
//
#include <hip/hip_runtime.h>

// ---------------------------------------------------------------------------
// TripletContrastiveLoss, fused for MI455X (gfx1250, wave32, WMMA).
//
// Roofline: GEMM f@f^T = 137 GFLOP on 33.5 MB (L2-resident) -> compute bound.
// v_wmma_f32_16x16x32_bf16 with bf16 hi/lo split (3 WMMAs ~ fp32 accuracy),
// fused distance+mask+min epilogue (268 MB distance matrix never exists).
// Round 2: global->LDS staging now uses gfx1250 async-copy
// (global_load_async_to_lds_b128, ASYNCcnt) with double-buffered LDS so the
// next tile streams in while WMMAs consume the current one; 1 barrier/step.
// ---------------------------------------------------------------------------

#define BDIM 8192
#define DDIM 1024
#define TM 128
#define TN 128
#define TK 32
#define LDSTRIDE 40   // shorts per LDS row: 80 B = 20 banks -> conflict-free b128
#define MARGIN 0.3f

typedef __attribute__((ext_vector_type(16))) __bf16 v16bf;
typedef __attribute__((ext_vector_type(8)))  float  v8f;

union FragAB { uint4 q[2]; v16bf v; };

__device__ __forceinline__ unsigned short f2bf(float x) {
  unsigned u = __float_as_uint(x);
  unsigned rnd = 0x7FFFu + ((u >> 16) & 1u);   // round-to-nearest-even
  return (unsigned short)((u + rnd) >> 16);
}
__device__ __forceinline__ float bf2f(unsigned short h) {
  return __uint_as_float(((unsigned)h) << 16);
}

// Async global->LDS 16-byte copy (CDNA5 path, tracked by ASYNCcnt).
// LDS address VGPR = low 32 bits of generic shared pointer (flat aperture:
// low dword is the LDS offset); global address = 64-bit VGPR pair, saddr=off.
__device__ __forceinline__ void async_b128(unsigned ldsaddr, unsigned long long gaddr) {
  asm volatile("global_load_async_to_lds_b128 %0, %1, off"
               :: "v"(ldsaddr), "v"(gaddr) : "memory");
}
__device__ __forceinline__ void wait_async0() {
  asm volatile("s_wait_asynccnt 0x0" ::: "memory");
}

// ---------------------------------------------------------------------------
// Kernel 1: L2-normalize each row, emit bf16 hi/lo decomposition.
// ---------------------------------------------------------------------------
__global__ __launch_bounds__(256)
void normalize_split_kernel(const float* __restrict__ F,
                            unsigned short* __restrict__ Fhi,
                            unsigned short* __restrict__ Flo) {
  __shared__ float red[8];
  const int row = blockIdx.x;
  const int t = threadIdx.x;
  const float4* src = reinterpret_cast<const float4*>(F + (size_t)row * DDIM);
  float4 x = src[t];
  float ss = x.x * x.x + x.y * x.y + x.z * x.z + x.w * x.w;
  #pragma unroll
  for (int m = 16; m >= 1; m >>= 1) ss += __shfl_xor(ss, m, 32);
  const int wave = t >> 5, lane = t & 31;
  if (lane == 0) red[wave] = ss;
  __syncthreads();
  float tot = red[0] + red[1] + red[2] + red[3] +
              red[4] + red[5] + red[6] + red[7];
  float inv = 1.0f / fmaxf(sqrtf(tot), 1e-12f);

  float y[4] = {x.x * inv, x.y * inv, x.z * inv, x.w * inv};
  unsigned short h[4], l[4];
  #pragma unroll
  for (int e = 0; e < 4; ++e) {
    h[e] = f2bf(y[e]);
    l[e] = f2bf(y[e] - bf2f(h[e]));   // residual term for bf16x3 GEMM
  }
  uint2 ph = make_uint2((unsigned)h[0] | ((unsigned)h[1] << 16),
                        (unsigned)h[2] | ((unsigned)h[3] << 16));
  uint2 pl = make_uint2((unsigned)l[0] | ((unsigned)l[1] << 16),
                        (unsigned)l[2] | ((unsigned)l[3] << 16));
  reinterpret_cast<uint2*>(Fhi + (size_t)row * DDIM)[t] = ph;
  reinterpret_cast<uint2*>(Flo + (size_t)row * DDIM)[t] = pl;
}

// ---------------------------------------------------------------------------
// Kernel 2: init per-row min buffers to +INF bits (ws poisoned; reset per call).
// ---------------------------------------------------------------------------
__global__ void init_minbuf(unsigned* __restrict__ posU,
                            unsigned* __restrict__ negU) {
  int i = blockIdx.x * blockDim.x + threadIdx.x;
  if (i < BDIM) { posU[i] = 0x7F800000u; negU[i] = 0x7F800000u; }
}

// ---------------------------------------------------------------------------
// One K-step of the bf16x3 GEMM: fragment loads (2x ds_read_b128 each, contiguous
// k order k = 16*half + s for A and B) + 24 v_wmma.
// ---------------------------------------------------------------------------
__device__ __forceinline__ void compute_step(
    const unsigned short* __restrict__ sAhi,
    const unsigned short* __restrict__ sAlo,
    const unsigned short* __restrict__ sBhi,
    const unsigned short* __restrict__ sBlo,
    int wm, int wn, int lrow, int kA0, int kA1, int kB0,
    v8f (&acc)[4][2]) {
  FragAB ahi[4], alo[4], bhi[2], blo[2];
  #pragma unroll
  for (int m = 0; m < 4; ++m) {
    const unsigned short* p = &sAhi[(wm * 64 + m * 16 + lrow) * LDSTRIDE];
    ahi[m].q[0] = *reinterpret_cast<const uint4*>(p + kA0);
    ahi[m].q[1] = *reinterpret_cast<const uint4*>(p + kA1);
    const unsigned short* q = &sAlo[(wm * 64 + m * 16 + lrow) * LDSTRIDE];
    alo[m].q[0] = *reinterpret_cast<const uint4*>(q + kA0);
    alo[m].q[1] = *reinterpret_cast<const uint4*>(q + kA1);
  }
  #pragma unroll
  for (int n = 0; n < 2; ++n) {
    const unsigned short* p = &sBhi[(wn * 32 + n * 16 + lrow) * LDSTRIDE + kB0];
    bhi[n].q[0] = *reinterpret_cast<const uint4*>(p);
    bhi[n].q[1] = *reinterpret_cast<const uint4*>(p + 8);
    const unsigned short* q = &sBlo[(wn * 32 + n * 16 + lrow) * LDSTRIDE + kB0];
    blo[n].q[0] = *reinterpret_cast<const uint4*>(q);
    blo[n].q[1] = *reinterpret_cast<const uint4*>(q + 8);
  }
  // bf16x3: s = hi*hi + hi*lo + lo*hi  (lo*lo ~ 2^-16, dropped)
  #pragma unroll
  for (int m = 0; m < 4; ++m)
    #pragma unroll
    for (int n = 0; n < 2; ++n) {
      acc[m][n] = __builtin_amdgcn_wmma_f32_16x16x32_bf16(
          false, ahi[m].v, false, bhi[n].v, (short)0, acc[m][n], false, false);
      acc[m][n] = __builtin_amdgcn_wmma_f32_16x16x32_bf16(
          false, ahi[m].v, false, blo[n].v, (short)0, acc[m][n], false, false);
      acc[m][n] = __builtin_amdgcn_wmma_f32_16x16x32_bf16(
          false, alo[m].v, false, bhi[n].v, (short)0, acc[m][n], false, false);
    }
}

// ---------------------------------------------------------------------------
// Kernel 3: fused 128x128-tile GEMM (bf16x3) + distance + mask + row-min.
// Double-buffered LDS fed by global_load_async_to_lds_b128 (ASYNCcnt).
// ---------------------------------------------------------------------------
__global__ __launch_bounds__(256)
void pairwise_min_kernel(const unsigned short* __restrict__ Fhi,
                         const unsigned short* __restrict__ Flo,
                         const int* __restrict__ labels,
                         const int* __restrict__ dom,
                         unsigned* __restrict__ posU,
                         unsigned* __restrict__ negU) {
  __shared__ unsigned short sA_hi[2][TM * LDSTRIDE];
  __shared__ unsigned short sA_lo[2][TM * LDSTRIDE];
  __shared__ unsigned short sB_hi[2][TN * LDSTRIDE];
  __shared__ unsigned short sB_lo[2][TN * LDSTRIDE];

  const int t     = threadIdx.x;
  const int wave  = t >> 5;
  const int lane  = t & 31;
  const int lrow  = lane & 15;   // M (A) / N (B,C) within 16x16 tile
  const int lhalf = lane >> 4;   // K-half (A,B) / M-offset-8 (C)
  const int wm = wave >> 2;      // 0..1
  const int wn = wave & 3;       // 0..3
  const int rowBlock = blockIdx.y * TM;
  const int colBlock = blockIdx.x * TN;

  // Cooperative tile loader mapping: thread -> (row, 16-short half), 32 B each
  const int ldRow  = t >> 1;
  const int ldHalf = t & 1;
  const size_t gA = (size_t)(rowBlock + ldRow) * DDIM + ldHalf * 16;
  const size_t gB = (size_t)(colBlock + ldRow) * DDIM + ldHalf * 16;
  const unsigned ldsByte = (unsigned)((ldRow * LDSTRIDE + ldHalf * 16) * 2);

  // Per-lane fragment K offsets (shorts) (contiguous k = 16*half + s, same for A and B):
  // A: lane<16 -> K {0..15};       lane>=16 -> K {16..31}
  // B: lane<16 -> K {0..15};       lane>=16 -> K {16..31}
  const int kA0 = lhalf ? 16 : 0;
  const int kA1 = lhalf ? 24 : 8;
  const int kB0 = lhalf * 16;

  // Issue the 8 async b128 copies (4 arrays x 32 B/thread) for one K-tile.
  auto issue_tile = [&](int buf, int kb) {
    unsigned long long a_hi = (unsigned long long)(size_t)(Fhi + gA + kb);
    unsigned long long a_lo = (unsigned long long)(size_t)(Flo + gA + kb);
    unsigned long long b_hi = (unsigned long long)(size_t)(Fhi + gB + kb);
    unsigned long long b_lo = (unsigned long long)(size_t)(Flo + gB + kb);
    unsigned dAhi = (unsigned)(size_t)&sA_hi[buf][0] + ldsByte;
    unsigned dAlo = (unsigned)(size_t)&sA_lo[buf][0] + ldsByte;
    unsigned dBhi = (unsigned)(size_t)&sB_hi[buf][0] + ldsByte;
    unsigned dBlo = (unsigned)(size_t)&sB_lo[buf][0] + ldsByte;
    async_b128(dAhi,      a_hi);      async_b128(dAhi + 16, a_hi + 8);
    async_b128(dAlo,      a_lo);      async_b128(dAlo + 16, a_lo + 8);
    async_b128(dBhi,      b_hi);      async_b128(dBhi + 16, b_hi + 8);
    async_b128(dBlo,      b_lo);      async_b128(dBlo + 16, b_lo + 8);
  };

  const v8f vzero = {0.f, 0.f, 0.f, 0.f, 0.f, 0.f, 0.f, 0.f};
  v8f acc[4][2];
  #pragma unroll
  for (int m = 0; m < 4; ++m)
    #pragma unroll
    for (int n = 0; n < 2; ++n) acc[m][n] = vzero;

  issue_tile(0, 0);   // prologue: stream first tile into buffer 0

  for (int kb = 0; kb < DDIM; kb += 2 * TK) {
    // ---- phase A: compute buf0 @ kb, prefetch kb+TK into buf1 ----
    wait_async0();        // this thread's buf0 data has landed in LDS
    __syncthreads();      // everyone's buf0 data visible; buf1 reads done
    if (kb + TK < DDIM) issue_tile(1, kb + TK);
    compute_step(sA_hi[0], sA_lo[0], sB_hi[0], sB_lo[0],
                 wm, wn, lrow, kA0, kA1, kB0, acc);
    // ---- phase B: compute buf1 @ kb+TK, prefetch kb+2*TK into buf0 ----
    wait_async0();
    __syncthreads();
    if (kb + 2 * TK < DDIM) issue_tile(0, kb + 2 * TK);
    compute_step(sA_hi[1], sA_lo[1], sB_hi[1], sB_lo[1],
                 wm, wn, lrow, kA0, kA1, kB0, acc);
  }

  // --- Fused epilogue: rows unit-norm => d^2 = 2 - 2*s ----------------------
  const float INFv = __uint_as_float(0x7F800000u);

  int labj[2]; bool fieldj[2];
  #pragma unroll
  for (int n = 0; n < 2; ++n) {
    const int j = colBlock + wn * 32 + n * 16 + lrow;
    labj[n]   = labels[j];
    fieldj[n] = (dom[j] == 1);
  }

  #pragma unroll
  for (int m = 0; m < 4; ++m) {
    #pragma unroll
    for (int r = 0; r < 8; ++r) {
      const int i = rowBlock + wm * 64 + m * 16 + lhalf * 8 + r;  // C layout
      const int  labi  = labels[i];
      const bool herbi = (dom[i] == 0);
      float pc = INFv, nc = INFv;
      #pragma unroll
      for (int n = 0; n < 2; ++n) {
        float s = acc[m][n][r];
        float d = sqrtf(fmaxf(2.0f - 2.0f * s, 0.0f));
        bool pair = herbi && fieldj[n];
        if (pair && (labi == labj[n])) pc = fminf(pc, d);
        if (pair && (labi != labj[n])) nc = fminf(nc, d);
      }
      // min across the 32 columns (2 per lane) held by this 16-lane group
      #pragma unroll
      for (int msk = 1; msk < 16; msk <<= 1) {
        pc = fminf(pc, __shfl_xor(pc, msk, 32));
        nc = fminf(nc, __shfl_xor(nc, msk, 32));
      }
      if (lrow == 0) {   // lanes 0 and 16: one writer per row
        // d >= 0 -> float order == uint bit order; atomic min exact/determ.
        if (pc < INFv) atomicMin(&posU[i], __float_as_uint(pc));
        if (nc < INFv) atomicMin(&negU[i], __float_as_uint(nc));
      }
    }
  }
}

// ---------------------------------------------------------------------------
// Kernel 4: hinge + masked mean over anchors.
// ---------------------------------------------------------------------------
__global__ __launch_bounds__(256)
void finalize_kernel(const unsigned* __restrict__ posU,
                     const unsigned* __restrict__ negU,
                     const int* __restrict__ dom,
                     float* __restrict__ out) {
  __shared__ float ssum[8];
  __shared__ float scnt[8];
  const int t = threadIdx.x;
  const float INFv = __uint_as_float(0x7F800000u);
  float sum = 0.f, cnt = 0.f;
  for (int i = t; i < BDIM; i += 256) {
    float p = __uint_as_float(posU[i]);
    float q = __uint_as_float(negU[i]);
    bool valid = (dom[i] == 0) && (p < INFv) && (q < INFv);
    if (valid) {
      sum += fmaxf(p - q + MARGIN, 0.0f);
      cnt += 1.0f;
    }
  }
  #pragma unroll
  for (int m = 16; m >= 1; m >>= 1) {
    sum += __shfl_xor(sum, m, 32);
    cnt += __shfl_xor(cnt, m, 32);
  }
  const int wave = t >> 5, lane = t & 31;
  if (lane == 0) { ssum[wave] = sum; scnt[wave] = cnt; }
  __syncthreads();
  if (t == 0) {
    float S = 0.f, C = 0.f;
    #pragma unroll
    for (int w = 0; w < 8; ++w) { S += ssum[w]; C += scnt[w]; }
    out[0] = (C > 0.f) ? (S / C) : 0.f;
  }
}

// ---------------------------------------------------------------------------
extern "C" void kernel_launch(void* const* d_in, const int* in_sizes, int n_in,
                              void* d_out, int out_size, void* d_ws, size_t ws_size,
                              hipStream_t stream) {
  const float* F    = (const float*)d_in[0];
  const int* labels = (const int*)d_in[1];
  const int* dom    = (const int*)d_in[2];

  // Workspace layout (~33.6 MB):
  unsigned short* Fhi = (unsigned short*)d_ws;               // 8192*1024 bf16
  unsigned short* Flo = Fhi + (size_t)BDIM * DDIM;           // 8192*1024 bf16
  unsigned* posU = (unsigned*)(Flo + (size_t)BDIM * DDIM);   // 8192 u32
  unsigned* negU = posU + BDIM;                              // 8192 u32

  normalize_split_kernel<<<BDIM, 256, 0, stream>>>(F, Fhi, Flo);
  init_minbuf<<<BDIM / 256, 256, 0, stream>>>(posU, negU);
  dim3 grid(BDIM / TN, BDIM / TM);   // 64 x 64 = 4096 blocks
  pairwise_min_kernel<<<grid, 256, 0, stream>>>(Fhi, Flo, labels, dom, posU, negU);
  finalize_kernel<<<1, 256, 0, stream>>>(posU, negU, dom, (float*)d_out);
}
